// Dynamic_DeeperGCN_24232205484263
// MI455X (gfx1250) — matrix operand order repaired and checked by reading, not compiled, not run on hardware
//
#include <hip/hip_runtime.h>
#include <hip/hip_bf16.h>
#include <cmath>

typedef _Float16 v16h __attribute__((ext_vector_type(16)));
typedef _Float16 v8h  __attribute__((ext_vector_type(8)));
typedef float    v8f  __attribute__((ext_vector_type(8)));

#define NB 32      // batch
#define NN 512     // nodes
#define ND 768     // feature dim
#define NL 4       // layers
#define BNROWS (NB * NN)   // 16384

__device__ __forceinline__ float sigmoidf_(float x) {
  return 1.0f / (1.0f + __expf(-x));
}

__device__ __forceinline__ v16h make_frag(const _Float16* p0, const _Float16* p1) {
  return __builtin_shufflevector(*(const v8h*)p0, *(const v8h*)p1,
                                 0,1,2,3,4,5,6,7,8,9,10,11,12,13,14,15);
}

enum { MODE_QK = 0, MODE_ADJ = 1, MODE_AGG = 2 };

// ---------------------------------------------------------------------------
// Batched WMMA GEMM: C[M,N] = A[M,K] * B[K,N], both operands f16 and both
// stored K-contiguous: A row-major [M][K], BT = B transposed [N][K].
// Block = 256 threads = 8 waves (2x4); block tile 64M x 256N; wave tile
// 32M x 64N = 8 v_wmma_f32_16x16x32_f16 per K-step. LDS double-buffered,
// software-pipelined global->reg->LDS staging, one barrier per K-step.
// ---------------------------------------------------------------------------
__global__ __launch_bounds__(256) void gemm_wmma_kernel(
    const _Float16* __restrict__ A, const _Float16* __restrict__ BT,
    int K, int ldA, int ldBT, size_t strideA, size_t strideBT,
    int mode, int layer,
    _Float16* __restrict__ outF16,
    float* __restrict__ outF32,
    float* __restrict__ adjsOut,
    const float* __restrict__ aux,   // pmask (ADJ) or dis (AGG)
    float scale)
{
  __shared__ __align__(16) _Float16 At[2][64][40];   // [buf][m-in-tile][k]
  __shared__ __align__(16) _Float16 Bt[2][256][40];  // [buf][n-in-tile][k]

  const int b    = blockIdx.z;
  const int m0   = blockIdx.y * 64;
  const int n0   = blockIdx.x * 256;
  const int tid  = threadIdx.x;
  const int lane = tid & 31;
  const int w    = tid >> 5;
  const int wr = w >> 2, wc = w & 3;
  const int mo = wr << 5;        // 0,32
  const int no = wc << 6;        // 0,64,128,192
  const int lr = lane & 15;
  const int hi = lane >> 4;      // K-half select (wave32 WMMA layout)

  // cooperative-load thread assignments (K-contiguous rows for both tiles)
  const int arow = tid >> 2;          // 0..63
  const int aseg = (tid & 3) << 3;    // 0,8,16,24
  const _Float16* Ab = A  + (size_t)b * strideA  + (size_t)(m0 + arow) * ldA + aseg;
  const _Float16* Bb = BT + (size_t)b * strideBT + (size_t)(n0 + tid) * ldBT;

  v8f acc[2][4];
  {
    v8f z = {};
    #pragma unroll
    for (int i = 0; i < 2; ++i)
      #pragma unroll
      for (int j = 0; j < 4; ++j) acc[i][j] = z;
  }

  const int steps = K >> 5;

  // ---- prologue: stage K-step 0 into buffer 0
  v8h a_reg  = *(const v8h*)Ab;
  v8h b_reg0 = *(const v8h*)(Bb + 0);
  v8h b_reg1 = *(const v8h*)(Bb + 8);
  v8h b_reg2 = *(const v8h*)(Bb + 16);
  v8h b_reg3 = *(const v8h*)(Bb + 24);
  *(v8h*)&At[0][arow][aseg] = a_reg;
  *(v8h*)&Bt[0][tid][0]  = b_reg0;
  *(v8h*)&Bt[0][tid][8]  = b_reg1;
  *(v8h*)&Bt[0][tid][16] = b_reg2;
  *(v8h*)&Bt[0][tid][24] = b_reg3;
  __syncthreads();

  for (int s = 0; s < steps; ++s) {
    const int cur = s & 1;
    const bool more = (s + 1) < steps;

    // pipeline: fetch next K-step into registers while computing current
    if (more) {
      const _Float16* ap = Ab + (size_t)(s + 1) * 32;
      const _Float16* bp = Bb + (size_t)(s + 1) * 32;
      a_reg  = *(const v8h*)ap;
      b_reg0 = *(const v8h*)(bp + 0);
      b_reg1 = *(const v8h*)(bp + 8);
      b_reg2 = *(const v8h*)(bp + 16);
      b_reg3 = *(const v8h*)(bp + 24);
    }
    if (s + 2 < steps) {
      __builtin_prefetch(Ab + (size_t)(s + 2) * 32, 0, 1);
      __builtin_prefetch(Bb + (size_t)(s + 2) * 32, 0, 1);
    }

    // A fragments: elems 0..7 -> K=hi*8+0..7, elems 8..15 -> K=hi*8+16..23
    const _Float16* ap0 = &At[cur][mo + lr][hi * 8];
    const _Float16* ap1 = &At[cur][mo + 16 + lr][hi * 8];
    v16h af0 = make_frag(ap0, ap0 + 16);
    v16h af1 = make_frag(ap1, ap1 + 16);

    #pragma unroll
    for (int ni = 0; ni < 4; ++ni) {
      // B fragment: elems 0..7 -> K=hi*8+0..7, elems 8..15 -> K=hi*8+16..23, col = no + ni*16 + lr
      const _Float16* bp = &Bt[cur][no + (ni << 4) + lr][hi * 8];
      v16h bf = make_frag(bp, bp + 16);
      acc[0][ni] = __builtin_amdgcn_wmma_f32_16x16x32_f16(false, af0, false, bf,
                                                          (short)0, acc[0][ni], false, false);
      acc[1][ni] = __builtin_amdgcn_wmma_f32_16x16x32_f16(false, af1, false, bf,
                                                          (short)0, acc[1][ni], false, false);
    }

    if (more) {
      const int nxt = cur ^ 1;
      *(v8h*)&At[nxt][arow][aseg] = a_reg;
      *(v8h*)&Bt[nxt][tid][0]  = b_reg0;
      *(v8h*)&Bt[nxt][tid][8]  = b_reg1;
      *(v8h*)&Bt[nxt][tid][16] = b_reg2;
      *(v8h*)&Bt[nxt][tid][24] = b_reg3;
    }
    __syncthreads();
  }

  // C/D layout: lane holds col = lane&15; VGPR j holds row = hi*8 + j
  const int growb = m0 + mo + (hi << 3);
  const int gcolb = n0 + no + lr;

  if (mode == MODE_QK) {
    #pragma unroll
    for (int mi = 0; mi < 2; ++mi)
      #pragma unroll
      for (int j = 0; j < 8; ++j) {
        size_t r = (size_t)(growb + mi * 16 + j) * ND;
        #pragma unroll
        for (int ni = 0; ni < 4; ++ni)
          outF16[r + gcolb + ni * 16] = (_Float16)acc[mi][ni][j];
      }
  } else if (mode == MODE_ADJ) {
    const float* pm = aux + b * NN;
    #pragma unroll
    for (int mi = 0; mi < 2; ++mi)
      #pragma unroll
      for (int j = 0; j < 8; ++j) {
        int n = growb + mi * 16 + j;
        float pn = pm[n];
        size_t base = ((size_t)b * NN + n) * NN;
        #pragma unroll
        for (int ni = 0; ni < 4; ++ni) {
          int m = gcolb + ni * 16;
          float v = sigmoidf_(acc[mi][ni][j] * scale) * pn * pm[m];
          outF32[base + m] = v;
          adjsOut[(base + m) * NL + layer] = v;   // adjs stacked on axis -1
        }
      }
  } else { // MODE_AGG
    const float* ds = aux + b * NN;
    #pragma unroll
    for (int mi = 0; mi < 2; ++mi)
      #pragma unroll
      for (int j = 0; j < 8; ++j) {
        int m = growb + mi * 16 + j;
        float dm = ds[m];
        size_t pbase = (((size_t)b * NN + m) * (NL + 1) + (layer + 1)) * ND;
        size_t fbase = ((size_t)b * NN + m) * ND;
        #pragma unroll
        for (int ni = 0; ni < 4; ++ni) {
          int d = gcolb + ni * 16;
          float v = acc[mi][ni][j] * dm;
          outF32[pbase + d] = v;
          outF16[fbase + d] = (_Float16)v;
        }
      }
  }
}

// ---------------------------------------------------------------------------
// Helper kernels
// ---------------------------------------------------------------------------
// weights: f32 [NL][ND][ND] -> f16 transposed per layer: dst[l][n][k] = src[l][k][n]
__global__ void trw_kernel(const float* __restrict__ src, _Float16* __restrict__ dst) {
  __shared__ float tile[32][33];
  int l = blockIdx.z, c0 = blockIdx.x * 32, r0 = blockIdx.y * 32;
  const float* s = src + (size_t)l * ND * ND;
  _Float16* d = dst + (size_t)l * ND * ND;
  for (int i = threadIdx.y; i < 32; i += 8)
    tile[i][threadIdx.x] = s[(size_t)(r0 + i) * ND + c0 + threadIdx.x];
  __syncthreads();
  for (int i = threadIdx.y; i < 32; i += 8)
    d[(size_t)(c0 + i) * ND + r0 + threadIdx.x] = (_Float16)tile[threadIdx.x][i];
}

__global__ void feat_init_kernel(const float* __restrict__ feature,
                                 _Float16* __restrict__ f16, float* __restrict__ preds) {
  size_t i = (size_t)blockIdx.x * blockDim.x + threadIdx.x;
  if (i >= (size_t)BNROWS * ND) return;
  float v = feature[i];
  f16[i] = (_Float16)v;
  size_t bn = i / ND, d = i % ND;
  preds[bn * (NL + 1) * ND + d] = v;   // preds[:, :, 0, :]
}

__global__ __launch_bounds__(256) void psum_kernel(const float* __restrict__ pmask,
                                                   float* __restrict__ p_sum) {
  int b = blockIdx.x, tid = threadIdx.x;
  __shared__ float red[256];
  red[tid] = pmask[b * NN + tid] + pmask[b * NN + 256 + tid];
  __syncthreads();
  for (int off = 128; off; off >>= 1) { if (tid < off) red[tid] += red[tid + off]; __syncthreads(); }
  if (tid == 0) p_sum[b] = red[0];
}

// f16 transpose: dst[b][c][r] = src[b][r][c]  (R=NN rows, C=ND cols)
__global__ void tr_feat_kernel(const _Float16* __restrict__ src, _Float16* __restrict__ dst) {
  __shared__ _Float16 tile[32][33];
  int b = blockIdx.z, c0 = blockIdx.x * 32, r0 = blockIdx.y * 32;
  const _Float16* s = src + (size_t)b * NN * ND;
  _Float16* d = dst + (size_t)b * ND * NN;
  for (int i = threadIdx.y; i < 32; i += 8)
    tile[i][threadIdx.x] = s[(size_t)(r0 + i) * ND + c0 + threadIdx.x];
  __syncthreads();
  for (int i = threadIdx.y; i < 32; i += 8)
    d[(size_t)(c0 + i) * NN + r0 + threadIdx.x] = tile[threadIdx.x][i];
}

// adjT16[b][m][n] = (f16)(adj[b][n][m] * dis[b][n])
__global__ void tr_adj_kernel(const float* __restrict__ adj, const float* __restrict__ dis,
                              _Float16* __restrict__ dst) {
  __shared__ float tile[32][33];
  int b = blockIdx.z, m0t = blockIdx.x * 32, n0t = blockIdx.y * 32;
  const float* s = adj + (size_t)b * NN * NN;
  const float* dv = dis + b * NN;
  _Float16* d = dst + (size_t)b * NN * NN;
  for (int i = threadIdx.y; i < 32; i += 8)
    tile[i][threadIdx.x] = s[(size_t)(n0t + i) * NN + m0t + threadIdx.x] * dv[n0t + i];
  __syncthreads();
  for (int i = threadIdx.y; i < 32; i += 8)
    d[(size_t)(m0t + i) * NN + n0t + threadIdx.x] = (_Float16)tile[threadIdx.x][i];
}

// one wave per adjacency row: deg, D^-1/2, row sums for sparsity stats
__global__ __launch_bounds__(256) void deg_kernel(const float* __restrict__ adj,
                                                  float* __restrict__ dis,
                                                  float* __restrict__ row_s,
                                                  float* __restrict__ row_rn) {
  int w = threadIdx.x >> 5, lane = threadIdx.x & 31;
  int row = blockIdx.x * 8 + w;
  const float* r = adj + (size_t)row * NN;
  float s = 0.f;
  for (int i = lane; i < NN; i += 32) s += r[i];
  #pragma unroll
  for (int off = 16; off; off >>= 1) s += __shfl_down(s, off, 32);
  if (lane == 0) {
    dis[row]    = (s > 0.f) ? rsqrtf(fmaxf(s, 1e-12f)) : 0.f;
    row_s[row]  = s;
    row_rn[row] = s - r[NN - 1];   // row sum excluding last column
  }
}

__global__ __launch_bounds__(256) void layer_stats_kernel(
    const float* __restrict__ row_s, const float* __restrict__ row_rn, int layer,
    float* __restrict__ c_acc, float* __restrict__ g_acc, float* __restrict__ l0_part) {
  int b = blockIdx.x, tid = threadIdx.x;
  __shared__ float rs[256], rg[256];
  float s = 0.f, g = 0.f;
  for (int i = tid; i < NN; i += 256) {
    int r = b * NN + i;
    s += row_s[r];
    if (i < NN - 1) g += row_rn[r];
  }
  rs[tid] = s; rg[tid] = g; __syncthreads();
  for (int off = 128; off; off >>= 1) {
    if (tid < off) { rs[tid] += rs[tid + off]; rg[tid] += rg[tid + off]; }
    __syncthreads();
  }
  if (tid == 0) {
    l0_part[layer * NB + b] = rs[0];
    g_acc[b * NL + layer]   = rg[0];
    c_acc[b * NL + layer]   = row_rn[b * NN + NN - 1];
  }
}

// attention pooling over preds[b,n,l,:]
__global__ __launch_bounds__(256) void pool_kernel(
    const float* __restrict__ preds, const float* __restrict__ proj_w,
    const float* __restrict__ proj_b, float* __restrict__ out, float* __restrict__ retain) {
  int bn = blockIdx.x, tid = threadIdx.x;
  const float* pp = preds + (size_t)bn * (NL + 1) * ND;
  float w0 = proj_w[tid], w1 = proj_w[tid + 256], w2 = proj_w[tid + 512];
  float o0 = 0.f, o1 = 0.f, o2 = 0.f;
  __shared__ float red[256];
  __shared__ float ret[NL + 1];
  float pb = proj_b[0];
  for (int l = 0; l <= NL; ++l) {
    const float* p = pp + (size_t)l * ND;
    float p0 = p[tid], p1 = p[tid + 256], p2 = p[tid + 512];
    red[tid] = p0 * w0 + p1 * w1 + p2 * w2;
    __syncthreads();
    for (int off = 128; off; off >>= 1) { if (tid < off) red[tid] += red[tid + off]; __syncthreads(); }
    if (tid == 0) ret[l] = 1.0f / (1.0f + __expf(-(red[0] + pb)));
    __syncthreads();
    float rl = ret[l];
    o0 += rl * p0; o1 += rl * p1; o2 += rl * p2;
  }
  size_t ob = (size_t)bn * ND;
  out[ob + tid] = o0; out[ob + 256 + tid] = o1; out[ob + 512 + tid] = o2;
  if (tid <= NL) retain[(size_t)bn * (NL + 1) + tid] = ret[tid];
}

__global__ void finalize_kernel(const float* __restrict__ p_sum,
                                const float* __restrict__ c_acc,
                                const float* __restrict__ g_acc,
                                const float* __restrict__ l0_part,
                                float* __restrict__ l0_out,
                                float* __restrict__ c_out, float* __restrict__ g_out) {
  int tid = threadIdx.x;
  if (tid == 0) {
    float fs = 0.f;
    for (int b = 0; b < NB; ++b) fs += p_sum[b] * p_sum[b];
    fs += 1e-8f;
    float t = 0.f;
    for (int i = 0; i < NL * NB; ++i) t += l0_part[i];
    l0_out[0] = (t / fs) / (float)NL;
  }
  if (tid < NB * NL) {
    int b = tid >> 2, l = tid & 3;
    float ps = p_sum[b];
    c_out[b * NL + l] = c_acc[b * NL + l] / ps;
    g_out[b * NL + l] = g_acc[b * NL + l] / (ps * ps);
  }
}

// ---------------------------------------------------------------------------
// Host orchestration
// ---------------------------------------------------------------------------
extern "C" void kernel_launch(void* const* d_in, const int* in_sizes, int n_in,
                              void* d_out, int out_size, void* d_ws, size_t ws_size,
                              hipStream_t stream) {
  (void)in_sizes; (void)n_in; (void)out_size; (void)ws_size;

  const float* pmask   = (const float*)d_in[0];
  const float* feature = (const float*)d_in[1];
  const float* Wq      = (const float*)d_in[2];
  const float* Wk      = (const float*)d_in[3];
  const float* proj_w  = (const float*)d_in[4];
  const float* proj_b  = (const float*)d_in[5];

  float* out_ptr    = (float*)d_out;
  float* retain_ptr = out_ptr + (size_t)NB * NN * ND;
  float* adjs_ptr   = retain_ptr + (size_t)NB * NN * (NL + 1);
  float* l0_ptr     = adjs_ptr + (size_t)NB * NN * NN * NL;
  float* c_ptr      = l0_ptr + 1;
  float* g_ptr      = c_ptr + NB * NL;

  // workspace bump allocator (256B aligned)
  char* ws = (char*)d_ws;
  size_t off = 0;
  auto alloc = [&](size_t bytes) -> void* {
    void* p = ws + off;
    off += (bytes + 255) & ~(size_t)255;
    return p;
  };
  _Float16* feat16a = (_Float16*)alloc((size_t)BNROWS * ND * 2);
  _Float16* feat16b = (_Float16*)alloc((size_t)BNROWS * ND * 2);
  _Float16* featT16 = (_Float16*)alloc((size_t)NB * ND * NN * 2);
  _Float16* wq16T   = (_Float16*)alloc((size_t)NL * ND * ND * 2);
  _Float16* wk16T   = (_Float16*)alloc((size_t)NL * ND * ND * 2);
  _Float16* q16     = (_Float16*)alloc((size_t)BNROWS * ND * 2);
  _Float16* k16     = (_Float16*)alloc((size_t)BNROWS * ND * 2);
  _Float16* adjT16  = (_Float16*)alloc((size_t)NB * NN * NN * 2);
  float*    adj_ws  = (float*)alloc((size_t)NB * NN * NN * 4);
  float*    preds   = (float*)alloc((size_t)BNROWS * (NL + 1) * ND * 4);
  float*    dis     = (float*)alloc((size_t)BNROWS * 4);
  float*    row_s   = (float*)alloc((size_t)BNROWS * 4);
  float*    row_rn  = (float*)alloc((size_t)BNROWS * 4);
  float*    p_sum   = (float*)alloc((size_t)NB * 4);
  float*    c_acc   = (float*)alloc((size_t)NB * NL * 4);
  float*    g_acc   = (float*)alloc((size_t)NB * NL * 4);
  float*    l0_part = (float*)alloc((size_t)NL * NB * 4);

  const float scale = 1.0f / sqrtf((float)ND);

  // one-time: transposed f16 weights, mask sums, initial feature / preds[:,:,0,:]
  trw_kernel<<<dim3(ND / 32, ND / 32, NL), dim3(32, 8), 0, stream>>>(Wq, wq16T);
  trw_kernel<<<dim3(ND / 32, ND / 32, NL), dim3(32, 8), 0, stream>>>(Wk, wk16T);
  psum_kernel<<<NB, 256, 0, stream>>>(pmask, p_sum);
  feat_init_kernel<<<(int)(((size_t)BNROWS * ND + 255) / 256), 256, 0, stream>>>(
      feature, feat16a, preds);

  for (int l = 0; l < NL; ++l) {
    _Float16* fin  = (l & 1) ? feat16b : feat16a;
    _Float16* fout = (l & 1) ? feat16a : feat16b;

    // feature^T per batch (B operand for AGG)
    tr_feat_kernel<<<dim3(ND / 32, NN / 32, NB), dim3(32, 8), 0, stream>>>(fin, featT16);

    // q = feature @ Wq[l], k = feature @ Wk[l]  (flattened M = B*N; BT = W^T)
    dim3 gQK(ND / 256, BNROWS / 64, 1);
    gemm_wmma_kernel<<<gQK, 256, 0, stream>>>(
        fin, wq16T + (size_t)l * ND * ND, ND, ND, ND, 0, 0,
        MODE_QK, l, q16, nullptr, nullptr, nullptr, 0.f);
    gemm_wmma_kernel<<<gQK, 256, 0, stream>>>(
        fin, wk16T + (size_t)l * ND * ND, ND, ND, ND, 0, 0,
        MODE_QK, l, k16, nullptr, nullptr, nullptr, 0.f);

    // adj = sigmoid(q k^T * scale) * full_adj ; BT for q@k^T is k16 itself
    dim3 gADJ(NN / 256, NN / 64, NB);
    gemm_wmma_kernel<<<gADJ, 256, 0, stream>>>(
        q16, k16, ND, ND, ND, (size_t)NN * ND, (size_t)NN * ND,
        MODE_ADJ, l, nullptr, adj_ws, adjs_ptr, pmask, scale);

    // degrees, D^-1/2, sparsity row sums
    deg_kernel<<<BNROWS / 8, 256, 0, stream>>>(adj_ws, dis, row_s, row_rn);
    layer_stats_kernel<<<NB, 256, 0, stream>>>(row_s, row_rn, l, c_acc, g_acc, l0_part);

    // adjT16[m][n] = adj[n][m]*dis[n]; feature' = dis[m] * (adjT16 @ feature)
    tr_adj_kernel<<<dim3(NN / 32, NN / 32, NB), dim3(32, 8), 0, stream>>>(adj_ws, dis, adjT16);
    dim3 gAGG(ND / 256, NN / 64, NB);
    gemm_wmma_kernel<<<gAGG, 256, 0, stream>>>(
        adjT16, featT16, NN, NN, NN, (size_t)NN * NN, (size_t)ND * NN,
        MODE_AGG, l, fout, preds, nullptr, dis, 0.f);
  }

  // attention pooling + scalar outputs
  pool_kernel<<<BNROWS, 256, 0, stream>>>(preds, proj_w, proj_b, out_ptr, retain_ptr);
  finalize_kernel<<<1, 128, 0, stream>>>(p_sum, c_acc, g_acc, l0_part, l0_ptr, c_ptr, g_ptr);
}
